// multiplicative_attention_model_33543694582002
// MI455X (gfx1250) — matrix operand order repaired and checked by reading, not compiled, not run on hardware
//
#include <hip/hip_runtime.h>
#include <math.h>

// ---------------------------------------------------------------------------
// Model dims
// ---------------------------------------------------------------------------
#define BB 16          // batch
#define TT 128         // seq len
#define HH 512         // hidden
#define DD 256         // embed dim
#define G4 2048        // 4*H (gate width)
#define VO 32001       // vocab out
#define VOPAD 32064    // padded to multiple of 64 (4-tile N groups)

typedef unsigned short bfu;  // bf16 storage bits
typedef __attribute__((ext_vector_type(16))) __bf16 v16bf;
typedef __attribute__((ext_vector_type(8)))  float  v8f;

union Frag { v16bf v; uint4 q[2]; };

// ---------------------------------------------------------------------------
// bf16 helpers (manual RNE so host never touches __bf16 values)
// ---------------------------------------------------------------------------
__device__ __forceinline__ bfu f2bf(float f) {
  unsigned int u = __float_as_uint(f);
  u += 0x7FFFu + ((u >> 16) & 1u);       // round-to-nearest-even
  return (bfu)(u >> 16);
}
__device__ __forceinline__ float bf2f(bfu h) {
  return __uint_as_float(((unsigned int)h) << 16);
}
__device__ __forceinline__ float sigm(float x) { return 1.0f / (1.0f + __expf(-x)); }

// ---------------------------------------------------------------------------
// WMMA fragment loaders.
// A (16xK row-major, [M][K]): lane L holds row L&15.
//   elements 0..7  <- K = kt*32 + (L>=16?8:0) + 0..7
//   elements 8..15 <- K = kt*32 + (L>=16?8:0) + 16..23
// B ([N][K] row-major, i.e. transposed weights): lane L holds col L&15.
//   elements 0..7  <- K = kt*32 + (L>=16?8:0) + 0..7, 8..15 <- +16..23 (same as A)
// ---------------------------------------------------------------------------
__device__ __forceinline__ v16bf load_a(const bfu* A, int lda, int mbase, int kt, int lane) {
  int row  = mbase + (lane & 15);
  int koff = kt * 32 + ((lane >> 4) ? 8 : 0);
  const uint4* p = (const uint4*)(A + (long long)row * lda + koff);
  Frag f; f.q[0] = p[0]; f.q[1] = p[2];
  return f.v;
}
__device__ __forceinline__ v16bf load_b(const bfu* BT, int K, int nbase, int kt, int lane) {
  int n    = nbase + (lane & 15);
  int koff = kt * 32 + ((lane >> 4) ? 8 : 0);
  const uint4* p = (const uint4*)(BT + (long long)n * K + koff);
  Frag f; f.q[0] = p[0]; f.q[1] = p[2];
  return f.v;
}
__device__ __forceinline__ v8f wmma_bf16(v16bf a, v16bf b, v8f c) {
  return __builtin_amdgcn_wmma_f32_16x16x32_bf16(false, a, false, b, (short)0, c, false, false);
}

// ---------------------------------------------------------------------------
// Weight convert + transpose: src f32 [K][N] -> dst bf16 [Npad][K] (zero pad)
// ---------------------------------------------------------------------------
__global__ void cvt_transpose(const float* __restrict__ src, bfu* __restrict__ dst,
                              int K, int N, int Npad) {
  long long total = (long long)Npad * K;
  for (long long i = (long long)blockIdx.x * blockDim.x + threadIdx.x; i < total;
       i += (long long)gridDim.x * blockDim.x) {
    int k = (int)(i % K);
    int n = (int)(i / K);
    float v = (n < N) ? src[(long long)k * N + n] : 0.0f;
    dst[i] = f2bf(v);
  }
}

// ---------------------------------------------------------------------------
// Embedding gather -> x bf16 [T*16 + b][D]  (time-major rows match z buffers)
// ---------------------------------------------------------------------------
__global__ void embed_kernel(const int* __restrict__ tokens, const float* __restrict__ emb,
                             bfu* __restrict__ x) {
  int i = blockIdx.x * blockDim.x + threadIdx.x;
  if (i >= BB * TT * DD) return;
  int d = i % DD;
  int r = i / DD;            // r = t*16 + b
  int b = r & 15;
  int t = r >> 4;
  int tok = tokens[b * TT + t];
  x[i] = f2bf(emb[(long long)tok * DD + d]);
}

// ---------------------------------------------------------------------------
// GEMM: C[M][ldc] f32 = A[M][K] bf16 @ BT[Npad][K] bf16 (+bias).
// Block = 128 threads = 4 waves. Register blocking: each wave computes
// 2 M-tiles x 4 N-tiles (32x64 output), 8 WMMAs per K-step from 12 b128
// loads (vs 4 loads/WMMA unblocked) -> ~2.7x less L2 operand traffic.
// Ntiles must be a multiple of 4 (N padded); store guards col < Nvalid.
// ---------------------------------------------------------------------------
__global__ __launch_bounds__(128) void gemm_bias_f32(
    const bfu* __restrict__ A, const bfu* __restrict__ BT, const float* __restrict__ bias,
    float* __restrict__ C, int K, int Nvalid, int Ntiles, int ldc) {
  int lane   = threadIdx.x & 31;
  int wave   = threadIdx.x >> 5;
  int mbase  = blockIdx.x * 32;          // two M tiles per wave
  int tgroup = blockIdx.y * 4 + wave;    // group of 4 consecutive N tiles
  if (tgroup * 4 >= Ntiles) return;
  int nbase0 = tgroup * 64;

  v8f acc[2][4] = {};
  int ktn = K >> 5;
  for (int kt = 0; kt < ktn; ++kt) {
    v16bf a0 = load_a(A, K, mbase, kt, lane);
    v16bf a1 = load_a(A, K, mbase + 16, kt, lane);
#pragma unroll
    for (int nt = 0; nt < 4; ++nt) {
      v16bf b = load_b(BT, K, nbase0 + nt * 16, kt, lane);
      acc[0][nt] = wmma_bf16(a0, b, acc[0][nt]);
      acc[1][nt] = wmma_bf16(a1, b, acc[1][nt]);
    }
  }
#pragma unroll
  for (int mt = 0; mt < 2; ++mt) {
    int rbase = mbase + mt * 16 + ((lane >> 4) ? 8 : 0);
#pragma unroll
    for (int nt = 0; nt < 4; ++nt) {
      int col = nbase0 + nt * 16 + (lane & 15);
      if (col < Nvalid) {
        float bv = bias ? bias[col] : 0.0f;
#pragma unroll
        for (int r = 0; r < 8; ++r)
          C[(long long)(rbase + r) * ldc + col] = acc[mt][nt][r] + bv;
      }
    }
  }
}

// ---------------------------------------------------------------------------
// Persistent LSTM layer: one workgroup, 16 waves.
// Per step: z = zin[t] (precomputed x@W+b) + h@U  via WMMA, then gates.
// Wave w owns N-tiles {32g + 16j + w} so i/f/g/o columns align per lane;
// c stays in registers, h lives in LDS as the next step's A matrix.
// ---------------------------------------------------------------------------
__global__ __launch_bounds__(512) void lstm_layer(
    const float* __restrict__ zin,   // [T*16][2048]
    const bfu*   __restrict__ UT,    // [2048][512]
    const float* __restrict__ h0, const float* __restrict__ c0,  // [16][512] or null
    bfu*   __restrict__ seq_out,     // [T*16][512]
    float* __restrict__ hT, float* __restrict__ cT) {            // [16][512] or null
  __shared__ __align__(16) bfu hbuf[BB * HH];
  int lane = threadIdx.x & 31;
  int wave = threadIdx.x >> 5;
  int lrow = lane & 15;
  int rhi  = (lane >> 4) ? 8 : 0;

  for (int i = threadIdx.x; i < BB * HH; i += 512)
    hbuf[i] = f2bf(h0 ? h0[i] : 0.0f);

  float cst[2][8];
#pragma unroll
  for (int j = 0; j < 2; ++j) {
    int col = j * 256 + wave * 16 + lrow;
#pragma unroll
    for (int r = 0; r < 8; ++r)
      cst[j][r] = c0 ? c0[(rhi + r) * HH + col] : 0.0f;
  }
  __syncthreads();

  for (int t = 0; t < TT; ++t) {
    v8f acc[8] = {};
    for (int kt = 0; kt < HH / 32; ++kt) {
      v16bf a = load_a(hbuf, HH, 0, kt, lane);
#pragma unroll
      for (int q = 0; q < 8; ++q) {
        int gate = q >> 1, j = q & 1;
        int ntile = gate * 32 + j * 16 + wave;
        acc[q] = wmma_bf16(a, load_b(UT, HH, ntile * 16, kt, lane), acc[q]);
      }
    }
    __syncthreads();  // all h reads complete before overwrite

    float hn[2][8];
#pragma unroll
    for (int j = 0; j < 2; ++j) {
      int col = j * 256 + wave * 16 + lrow;
#pragma unroll
      for (int r = 0; r < 8; ++r) {
        int row = rhi + r;
        const float* zr = zin + (long long)(t * 16 + row) * G4;
        float zi = acc[0 + j][r] + zr[0 * HH + col];
        float zf = acc[2 + j][r] + zr[1 * HH + col];
        float zg = acc[4 + j][r] + zr[2 * HH + col];
        float zo = acc[6 + j][r] + zr[3 * HH + col];
        float c  = sigm(zf) * cst[j][r] + sigm(zi) * tanhf(zg);
        float h  = sigm(zo) * tanhf(c);
        cst[j][r] = c;
        hn[j][r]  = h;
      }
    }
#pragma unroll
    for (int j = 0; j < 2; ++j) {
      int col = j * 256 + wave * 16 + lrow;
#pragma unroll
      for (int r = 0; r < 8; ++r) {
        int row = rhi + r;
        bfu hb = f2bf(hn[j][r]);
        hbuf[row * HH + col] = hb;
        seq_out[(long long)(t * 16 + row) * HH + col] = hb;
      }
    }
    __syncthreads();
  }

  if (hT) {
#pragma unroll
    for (int j = 0; j < 2; ++j) {
      int col = j * 256 + wave * 16 + lrow;
#pragma unroll
      for (int r = 0; r < 8; ++r) {
        hT[(rhi + r) * HH + col] = bf2f(hbuf[(rhi + r) * HH + col]);
        cT[(rhi + r) * HH + col] = cst[j][r];
      }
    }
  }
}

// ---------------------------------------------------------------------------
// Decoder: 2 LSTM cells + Luong dot attention, one persistent workgroup.
// Cell A z-input precomputed (seq2@Wd1+bd1); cell B does hA@Wd2 + hB@Ud2
// via WMMA each step. Attention per-batch softmax handled one-wave-per-b.
// Writes dec rows [b*T + t][ hB(512) | ctx(512) ] bf16.
// ---------------------------------------------------------------------------
__global__ __launch_bounds__(512) void decoder_kernel(
    const float* __restrict__ zinA,   // [T*16][2048]
    const bfu* __restrict__ Ud1T, const bfu* __restrict__ Wd2T, const bfu* __restrict__ Ud2T,
    const float* __restrict__ bd2,    // [2048]
    const float* __restrict__ h0A, const float* __restrict__ c0A,
    const float* __restrict__ h0B, const float* __restrict__ c0B,
    const bfu* __restrict__ seq2,     // [T*16][512]
    bfu* __restrict__ dec) {          // [B*T][1024]
  __shared__ __align__(16) bfu hA[BB * HH];
  __shared__ __align__(16) bfu hB[BB * HH];
  __shared__ float sc[BB * TT];
  int lane = threadIdx.x & 31;
  int wave = threadIdx.x >> 5;
  int lrow = lane & 15;
  int rhi  = (lane >> 4) ? 8 : 0;

  for (int i = threadIdx.x; i < BB * HH; i += 512) {
    hA[i] = f2bf(h0A[i]);
    hB[i] = f2bf(h0B[i]);
  }
  float cA[2][8], cB[2][8];
#pragma unroll
  for (int j = 0; j < 2; ++j) {
    int col = j * 256 + wave * 16 + lrow;
#pragma unroll
    for (int r = 0; r < 8; ++r) {
      cA[j][r] = c0A[(rhi + r) * HH + col];
      cB[j][r] = c0B[(rhi + r) * HH + col];
    }
  }
  __syncthreads();

  for (int t = 0; t < TT; ++t) {
    // ---- cell A: z = zinA[t] + hA@Ud1 ----
    {
      v8f acc[8] = {};
      for (int kt = 0; kt < HH / 32; ++kt) {
        v16bf a = load_a(hA, HH, 0, kt, lane);
#pragma unroll
        for (int q = 0; q < 8; ++q) {
          int gate = q >> 1, j = q & 1;
          int ntile = gate * 32 + j * 16 + wave;
          acc[q] = wmma_bf16(a, load_b(Ud1T, HH, ntile * 16, kt, lane), acc[q]);
        }
      }
      __syncthreads();
#pragma unroll
      for (int j = 0; j < 2; ++j) {
        int col = j * 256 + wave * 16 + lrow;
#pragma unroll
        for (int r = 0; r < 8; ++r) {
          int row = rhi + r;
          const float* zr = zinA + (long long)(t * 16 + row) * G4;
          float zi = acc[0 + j][r] + zr[0 * HH + col];
          float zf = acc[2 + j][r] + zr[1 * HH + col];
          float zg = acc[4 + j][r] + zr[2 * HH + col];
          float zo = acc[6 + j][r] + zr[3 * HH + col];
          float c  = sigm(zf) * cA[j][r] + sigm(zi) * tanhf(zg);
          float h  = sigm(zo) * tanhf(c);
          cA[j][r] = c;
          hA[row * HH + col] = f2bf(h);
        }
      }
      __syncthreads();
    }
    // ---- cell B: z = bd2 + hA@Wd2 + hB@Ud2 ----
    {
      v8f acc[8] = {};
      for (int kt = 0; kt < HH / 32; ++kt) {
        v16bf aA = load_a(hA, HH, 0, kt, lane);
        v16bf aB = load_a(hB, HH, 0, kt, lane);
#pragma unroll
        for (int q = 0; q < 8; ++q) {
          int gate = q >> 1, j = q & 1;
          int ntile = gate * 32 + j * 16 + wave;
          acc[q] = wmma_bf16(aA, load_b(Wd2T, HH, ntile * 16, kt, lane), acc[q]);
          acc[q] = wmma_bf16(aB, load_b(Ud2T, HH, ntile * 16, kt, lane), acc[q]);
        }
      }
      __syncthreads();
#pragma unroll
      for (int j = 0; j < 2; ++j) {
        int col = j * 256 + wave * 16 + lrow;
#pragma unroll
        for (int r = 0; r < 8; ++r) {
          int row = rhi + r;
          float zi = acc[0 + j][r] + bd2[0 * HH + col];
          float zf = acc[2 + j][r] + bd2[1 * HH + col];
          float zg = acc[4 + j][r] + bd2[2 * HH + col];
          float zo = acc[6 + j][r] + bd2[3 * HH + col];
          float c  = sigm(zf) * cB[j][r] + sigm(zi) * tanhf(zg);
          float h  = sigm(zo) * tanhf(c);
          cB[j][r] = c;
          hB[row * HH + col] = f2bf(h);
        }
      }
      __syncthreads();
    }
    // ---- attention scores: sc[b][s] = seq2[b,s,:] . hB[b,:] ----
#pragma unroll
    for (int rep = 0; rep < 4; ++rep) {
      int ii = threadIdx.x + rep * 512;  // 2048 (b,s) pairs
      int b = ii >> 7, s = ii & 127;
      const bfu* sp = seq2 + (long long)(s * 16 + b) * HH;
      const bfu* hp = hB + b * HH;
      float d = 0.0f;
      for (int h = 0; h < HH; ++h) d += bf2f(sp[h]) * bf2f(hp[h]);
      sc[b * TT + s] = d;
    }
    __syncthreads();
    // ---- softmax over s, wave w handles b = w ----
    {
      int b = wave;
      float v[4], m = -1e30f;
#pragma unroll
      for (int k = 0; k < 4; ++k) {
        v[k] = sc[b * TT + lane + 32 * k];
        m = fmaxf(m, v[k]);
      }
      for (int off = 16; off; off >>= 1) m = fmaxf(m, __shfl_xor(m, off, 32));
      float e[4], sum = 0.0f;
#pragma unroll
      for (int k = 0; k < 4; ++k) { e[k] = __expf(v[k] - m); sum += e[k]; }
      for (int off = 16; off; off >>= 1) sum += __shfl_xor(sum, off, 32);
      float inv = 1.0f / sum;
#pragma unroll
      for (int k = 0; k < 4; ++k) sc[b * TT + lane + 32 * k] = e[k] * inv;
    }
    __syncthreads();
    // ---- context + output row: wave w handles b = w ----
    {
      int b = wave;
      bfu* drow = dec + ((long long)b * TT + t) * 1024;
      for (int hh = lane; hh < HH; hh += 32) {
        float cx = 0.0f;
        for (int s = 0; s < TT; ++s)
          cx += sc[b * TT + s] * bf2f(seq2[(long long)(s * 16 + b) * HH + hh]);
        drow[hh]       = hB[b * HH + hh];
        drow[512 + hh] = f2bf(cx);
      }
    }
    __syncthreads();
  }
}

// ---------------------------------------------------------------------------
// Row softmax over d_out: 2048 rows x 32001, in place.
// ---------------------------------------------------------------------------
__global__ __launch_bounds__(256) void softmax_rows(float* __restrict__ out) {
  int row = blockIdx.x;
  float* p = out + (long long)row * VO;
  __shared__ float red[256];
  float m = -1e30f;
  for (int i = threadIdx.x; i < VO; i += 256) m = fmaxf(m, p[i]);
  red[threadIdx.x] = m;
  __syncthreads();
  for (int s = 128; s; s >>= 1) {
    if (threadIdx.x < s) red[threadIdx.x] = fmaxf(red[threadIdx.x], red[threadIdx.x + s]);
    __syncthreads();
  }
  m = red[0];
  __syncthreads();
  float sum = 0.0f;
  for (int i = threadIdx.x; i < VO; i += 256) {
    float e = __expf(p[i] - m);
    p[i] = e;
    sum += e;
  }
  red[threadIdx.x] = sum;
  __syncthreads();
  for (int s = 128; s; s >>= 1) {
    if (threadIdx.x < s) red[threadIdx.x] += red[threadIdx.x + s];
    __syncthreads();
  }
  float inv = 1.0f / red[0];
  for (int i = threadIdx.x; i < VO; i += 256) p[i] *= inv;
}

// ---------------------------------------------------------------------------
// Launch
// ---------------------------------------------------------------------------
extern "C" void kernel_launch(void* const* d_in, const int* in_sizes, int n_in,
                              void* d_out, int out_size, void* d_ws, size_t ws_size,
                              hipStream_t stream) {
  (void)in_sizes; (void)n_in; (void)out_size; (void)ws_size;
  const int*   tokens = (const int*)d_in[0];
  const float* emb = (const float*)d_in[1];
  const float* W1  = (const float*)d_in[2];
  const float* U1  = (const float*)d_in[3];
  const float* b1  = (const float*)d_in[4];
  const float* W2  = (const float*)d_in[5];
  const float* U2  = (const float*)d_in[6];
  const float* b2  = (const float*)d_in[7];
  const float* Wd1 = (const float*)d_in[8];
  const float* Ud1 = (const float*)d_in[9];
  const float* bd1 = (const float*)d_in[10];
  const float* Wd2 = (const float*)d_in[11];
  const float* Ud2 = (const float*)d_in[12];
  const float* bd2 = (const float*)d_in[13];
  const float* Wo  = (const float*)d_in[14];
  const float* bo  = (const float*)d_in[15];
  float* out = (float*)d_out;

  char* ws = (char*)d_ws;
  size_t off = 0;
  auto alloc = [&](size_t bytes) -> void* {
    void* p = ws + off;
    off += (bytes + 255) & ~(size_t)255;
    return p;
  };
  bfu* xbf  = (bfu*)alloc((size_t)2048 * 256 * 2);
  bfu* W1T  = (bfu*)alloc((size_t)2048 * 256 * 2);
  bfu* U1T  = (bfu*)alloc((size_t)2048 * 512 * 2);
  bfu* W2T  = (bfu*)alloc((size_t)2048 * 512 * 2);
  bfu* U2T  = (bfu*)alloc((size_t)2048 * 512 * 2);
  bfu* Wd1T = (bfu*)alloc((size_t)2048 * 512 * 2);
  bfu* Ud1T = (bfu*)alloc((size_t)2048 * 512 * 2);
  bfu* Wd2T = (bfu*)alloc((size_t)2048 * 512 * 2);
  bfu* Ud2T = (bfu*)alloc((size_t)2048 * 512 * 2);
  bfu* WoT  = (bfu*)alloc((size_t)VOPAD * 1024 * 2);
  float* zin  = (float*)alloc((size_t)2048 * 2048 * 4);
  bfu* seq1 = (bfu*)alloc((size_t)2048 * 512 * 2);
  bfu* seq2 = (bfu*)alloc((size_t)2048 * 512 * 2);
  bfu* dec  = (bfu*)alloc((size_t)2048 * 1024 * 2);
  float* h1 = (float*)alloc((size_t)16 * 512 * 4);
  float* c1 = (float*)alloc((size_t)16 * 512 * 4);
  float* h2 = (float*)alloc((size_t)16 * 512 * 4);
  float* c2 = (float*)alloc((size_t)16 * 512 * 4);

  // Weight conversion / transposition (bf16, [N][K] layout for WMMA B-frags)
  cvt_transpose<<<2048, 256, 0, stream>>>(W1,  W1T,  256, 2048, 2048);
  cvt_transpose<<<2048, 256, 0, stream>>>(U1,  U1T,  512, 2048, 2048);
  cvt_transpose<<<2048, 256, 0, stream>>>(W2,  W2T,  512, 2048, 2048);
  cvt_transpose<<<2048, 256, 0, stream>>>(U2,  U2T,  512, 2048, 2048);
  cvt_transpose<<<2048, 256, 0, stream>>>(Wd1, Wd1T, 512, 2048, 2048);
  cvt_transpose<<<2048, 256, 0, stream>>>(Ud1, Ud1T, 512, 2048, 2048);
  cvt_transpose<<<2048, 256, 0, stream>>>(Wd2, Wd2T, 512, 2048, 2048);
  cvt_transpose<<<2048, 256, 0, stream>>>(Ud2, Ud2T, 512, 2048, 2048);
  cvt_transpose<<<4096, 256, 0, stream>>>(Wo,  WoT,  1024, VO, VOPAD);

  embed_kernel<<<(BB * TT * DD + 255) / 256, 256, 0, stream>>>(tokens, emb, xbf);

  // Encoder layer 1  (M=2048 -> 64 blocks of 32 rows; N=2048 -> 128 tiles, 32 groups)
  gemm_bias_f32<<<dim3(64, 8), 128, 0, stream>>>(xbf, W1T, b1, zin, 256, 2048, 128, 2048);
  lstm_layer<<<1, 512, 0, stream>>>(zin, U1T, nullptr, nullptr, seq1, h1, c1);
  // Encoder layer 2 (init with layer-1 finals)
  gemm_bias_f32<<<dim3(64, 8), 128, 0, stream>>>(seq1, W2T, b2, zin, 512, 2048, 128, 2048);
  lstm_layer<<<1, 512, 0, stream>>>(zin, U2T, h1, c1, seq2, h2, c2);
  // Decoder (cellA init h1/c1, cellB init h2/c2)
  gemm_bias_f32<<<dim3(64, 8), 128, 0, stream>>>(seq2, Wd1T, bd1, zin, 512, 2048, 128, 2048);
  decoder_kernel<<<1, 512, 0, stream>>>(zin, Ud1T, Wd2T, Ud2T, bd2, h1, c1, h2, c2, seq2, dec);
  // Vocab projection (dominant GEMM): Ntiles = 32064/16 = 2004 (divisible by 4),
  // grid.y = ceil(501 groups / 4 waves) = 126, store-guarded to 32001.
  gemm_bias_f32<<<dim3(64, 126), 128, 0, stream>>>(dec, WoT, bo, out, 1024, VO, VOPAD / 16, VO);
  softmax_rows<<<2048, 256, 0, stream>>>(out);
}
